// GNN_Model_68908455297098
// MI455X (gfx1250) — compile-verified, matrix operand order checked by reading, not run on hardware
//
#include <hip/hip_runtime.h>
#include <hip/hip_bf16.h>

typedef __attribute__((ext_vector_type(16))) __bf16 v16bf;
typedef __attribute__((ext_vector_type(4)))  __bf16 v4bf;
typedef __attribute__((ext_vector_type(8)))  float  v8f;
typedef __attribute__((ext_vector_type(4)))  int    i32x4;
typedef __attribute__((ext_vector_type(2)))  int    i32x2;

#define BB   16
#define NN   1024
#define LL   4
#define FIN_ 56
#define DD   140
#define DP   160          // D padded to multiple of 32
#define KE   64           // F_IN padded
#define DFC_ 128
#define FINAL_ 256
#define BN   (BB*NN)

#define TKP  40           // LDS k-stride (32 + 8 pad): 80B rows, 16B aligned

// --- CDNA5 async global->LDS staging (ASYNCcnt path), probe-guarded --------
#if defined(__has_builtin)
#if __has_builtin(__builtin_amdgcn_global_load_async_to_lds_b128) && \
    __has_builtin(__builtin_amdgcn_global_load_async_to_lds_b64)
#define HAVE_ASYNC_LDS 1
#endif
#endif
#ifndef HAVE_ASYNC_LDS
#define HAVE_ASYNC_LDS 0
#endif

#define AS1 __attribute__((address_space(1)))
#define AS3 __attribute__((address_space(3)))

#if HAVE_ASYNC_LDS
__device__ __forceinline__ void async_lds_b128(const void* g, void* l) {
    __builtin_amdgcn_global_load_async_to_lds_b128(
        (AS1 i32x4*)(AS1 void*)g, (AS3 i32x4*)(AS3 void*)l, 0, 0);
}
__device__ __forceinline__ void async_lds_b64(const void* g, void* l) {
    __builtin_amdgcn_global_load_async_to_lds_b64(
        (AS1 i32x2*)(AS1 void*)g, (AS3 i32x2*)(AS3 void*)l, 0, 0);
}
#endif

__device__ __forceinline__ void wait_asynccnt0() {
#if HAVE_ASYNC_LDS
#if __has_builtin(__builtin_amdgcn_s_wait_asynccnt)
    __builtin_amdgcn_s_wait_asynccnt(0);
#else
    asm volatile("s_wait_asynccnt 0x0" ::: "memory");
#endif
#endif
}

// ---------------------------------------------------------------------------
// pad + cast helpers
// ---------------------------------------------------------------------------
__global__ void pad_cast_kernel(const float* __restrict__ src, __bf16* __restrict__ dst,
                                int rows, int sc, int dc) {
    int e = blockIdx.x * 256 + threadIdx.x;
    if (e >= rows * dc) return;
    int r = e / dc, c = e % dc;
    dst[e] = (c < sc) ? (__bf16)src[r * sc + c] : (__bf16)0.0f;
}

// dst[dRows][dCols] = src^T zero-padded ; src is [sIn][sOut] row-major
__global__ void pad_castT_kernel(const float* __restrict__ src, __bf16* __restrict__ dst,
                                 int sIn, int sOut, int dRows, int dCols) {
    int e = blockIdx.x * 256 + threadIdx.x;
    if (e >= dRows * dCols) return;
    int o = e / dCols, i = e % dCols;
    dst[e] = (o < sOut && i < sIn) ? (__bf16)src[i * sOut + o] : (__bf16)0.0f;
}

__global__ void pad_bias_kernel(const float* __restrict__ src, float* __restrict__ dst,
                                int n, int np) {
    int i = blockIdx.x * 256 + threadIdx.x;
    if (i < np) dst[i] = (i < n) ? src[i] : 0.0f;
}

// ---------------------------------------------------------------------------
// fragment k-index for CDNA5 16-bit 16x32 operand layout
// ---------------------------------------------------------------------------
__device__ __forceinline__ int frag_k(int e, int khi) {
    return e + ((e & 8) ? 8 : 0) + khi;
}

// ---------------------------------------------------------------------------
// WMMA bf16 GEMM:  C[m,n] = sum_k opA(A)[m,k] * opB(B)[n,k]
// Block tile 128(M) x 32(N) x 32(K), 8 waves = 4(wm) x 2(wn); each wave owns
// two 16-row subtiles sharing one B fragment -> 2 v_wmma per wave per k-step.
// Non-transposed tiles staged via GLOBAL_LOAD_ASYNC_TO_LDS when available.
// ---------------------------------------------------------------------------
template<bool OPA_T, bool OPB_T>
__global__ __launch_bounds__(256)
void gemm_nt_kernel(const __bf16* __restrict__ A, const __bf16* __restrict__ Bm,
                    float* __restrict__ Cf, __bf16* __restrict__ Cb,
                    const float* __restrict__ bias,
                    int M, int N, int K, int lda, int ldb, int ldc, int relu,
                    long long strideA, long long strideB, long long strideC) {
    __shared__ __bf16 As[128][TKP];
    __shared__ __bf16 Bs[32][TKP];

    int z = blockIdx.z;
    A  += (long long)z * strideA;
    Bm += (long long)z * strideB;
    Cf  = Cf ? Cf + (long long)z * strideC : Cf;
    Cb  = Cb ? Cb + (long long)z * strideC : Cb;

    int m0 = blockIdx.y * 128;
    int n0 = blockIdx.x * 32;
    int tid = threadIdx.x;
    int wave = tid >> 5, lane = tid & 31;
    int wm = wave & 3, wn = wave >> 2;

    v8f acc0 = {}, acc1 = {};

    for (int kb = 0; kb < K; kb += 32) {
        // ---- stage A tile (128x32)
        if constexpr (!OPA_T) {
#if HAVE_ASYNC_LDS
#pragma unroll
            for (int p = 0; p < 2; ++p) {
                int idx = tid * 8 + p * 2048;
                int m = idx >> 5, k = idx & 31;          // 8 consecutive k (16B)
                async_lds_b128(A + (m0 + m) * lda + kb + k, &As[m][k]);
            }
#else
#pragma unroll
            for (int p = 0; p < 4; ++p) {
                int idx = tid * 4 + p * 1024;
                int m = idx >> 5, k = idx & 31;
                *(v4bf*)&As[m][k] = *(const v4bf*)(A + (m0 + m) * lda + kb + k);
            }
#endif
        } else {
#pragma unroll
            for (int p = 0; p < 4; ++p) {
                int idx = tid * 4 + p * 1024;
                int k = idx >> 7, m = idx & 127;          // 4 consecutive m
                v4bf v = *(const v4bf*)(A + (kb + k) * lda + (m0 + m));
#pragma unroll
                for (int j = 0; j < 4; ++j) As[m + j][k] = v[j];
            }
        }
        // ---- stage B tile (32x32)
        if constexpr (!OPB_T) {
            int idx = tid * 4;
            int n = idx >> 5, k = idx & 31;               // 4 consecutive k (8B)
#if HAVE_ASYNC_LDS
            async_lds_b64(Bm + (n0 + n) * ldb + kb + k, &Bs[n][k]);
#else
            *(v4bf*)&Bs[n][k] = *(const v4bf*)(Bm + (n0 + n) * ldb + kb + k);
#endif
        } else {
            int idx = tid * 4;
            int k = idx >> 5, n = idx & 31;
            v4bf v = *(const v4bf*)(Bm + (kb + k) * ldb + (n0 + n));
#pragma unroll
            for (int j = 0; j < 4; ++j) Bs[n + j][k] = v[j];
        }
        if constexpr (!OPA_T || !OPB_T) wait_asynccnt0();
        __syncthreads();

        int ar = wm * 32 + (lane & 15);
        int br = wn * 16 + (lane & 15);
        int khi = (lane >> 4) * 8;
        v16bf a0, a1, bf;
#pragma unroll
        for (int e = 0; e < 16; ++e) {
            int k = frag_k(e, khi);
            a0[e] = As[ar][k];
            a1[e] = As[ar + 16][k];
            bf[e] = Bs[br][k];
        }
        acc0 = __builtin_amdgcn_wmma_f32_16x16x32_bf16(false, a0, false, bf,
                                                       (short)0, acc0, false, false);
        acc1 = __builtin_amdgcn_wmma_f32_16x16x32_bf16(false, a1, false, bf,
                                                       (short)0, acc1, false, false);
        __syncthreads();
    }

    int n = n0 + wn * 16 + (lane & 15);
    int mb = m0 + wm * 32 + ((lane >> 4) ? 8 : 0);
    float bv = bias ? bias[n] : 0.0f;
#pragma unroll
    for (int r = 0; r < 8; ++r) {
        float v0 = acc0[r] + bv;
        float v1 = acc1[r] + bv;
        if (relu) { v0 = fmaxf(v0, 0.0f); v1 = fmaxf(v1, 0.0f); }
        int i0 = (mb + r) * ldc + n;
        int i1 = (mb + 16 + r) * ldc + n;
        if (Cf) { Cf[i0] = v0; Cf[i1] = v1; }
        if (Cb) { Cb[i0] = (__bf16)v0; Cb[i1] = (__bf16)v1; }
    }
}

// ---------------------------------------------------------------------------
// Symmetric attention-logit GEMM: E = hA.h2^T + h2.hA^T  (per batch z)
// Block tile 128x32; 4 v_wmma per wave per k-step; async-staged tiles.
// ---------------------------------------------------------------------------
__global__ __launch_bounds__(256)
void gemm_sym_kernel(const __bf16* __restrict__ hA, const __bf16* __restrict__ h2,
                     float* __restrict__ E, int Kp) {
    __shared__ __bf16 AsA[128][TKP], AsH[128][TKP];
    __shared__ __bf16 BsH[32][TKP],  BsA[32][TKP];

    int z = blockIdx.z;
    const __bf16* pA = hA + (long long)z * NN * Kp;
    const __bf16* pH = h2 + (long long)z * NN * Kp;
    float* pE = E + (long long)z * NN * NN;

    int m0 = blockIdx.y * 128, n0 = blockIdx.x * 32;
    int tid = threadIdx.x, wave = tid >> 5, lane = tid & 31;
    int wm = wave & 3, wn = wave >> 2;

    v8f acc0 = {}, acc1 = {};
    for (int kb = 0; kb < Kp; kb += 32) {
#if HAVE_ASYNC_LDS
#pragma unroll
        for (int p = 0; p < 2; ++p) {
            int idx = tid * 8 + p * 2048;
            int m = idx >> 5, k = idx & 31;
            int off = (m0 + m) * Kp + kb + k;
            async_lds_b128(pA + off, &AsA[m][k]);
            async_lds_b128(pH + off, &AsH[m][k]);
        }
        {
            int idx = tid * 4;
            int n = idx >> 5, k = idx & 31;
            int off = (n0 + n) * Kp + kb + k;
            async_lds_b64(pH + off, &BsH[n][k]);
            async_lds_b64(pA + off, &BsA[n][k]);
        }
        wait_asynccnt0();
#else
#pragma unroll
        for (int p = 0; p < 4; ++p) {
            int idx = tid * 4 + p * 1024;
            int m = idx >> 5, k = idx & 31;
            int off = (m0 + m) * Kp + kb + k;
            *(v4bf*)&AsA[m][k] = *(const v4bf*)(pA + off);
            *(v4bf*)&AsH[m][k] = *(const v4bf*)(pH + off);
        }
        {
            int idx = tid * 4;
            int n = idx >> 5, k = idx & 31;
            int off = (n0 + n) * Kp + kb + k;
            *(v4bf*)&BsH[n][k] = *(const v4bf*)(pH + off);
            *(v4bf*)&BsA[n][k] = *(const v4bf*)(pA + off);
        }
#endif
        __syncthreads();

        int ar = wm * 32 + (lane & 15);
        int br = wn * 16 + (lane & 15);
        int khi = (lane >> 4) * 8;
        v16bf aA0, aA1, aH0, aH1, bH, bA;
#pragma unroll
        for (int e = 0; e < 16; ++e) {
            int k = frag_k(e, khi);
            aA0[e] = AsA[ar][k];      aA1[e] = AsA[ar + 16][k];
            aH0[e] = AsH[ar][k];      aH1[e] = AsH[ar + 16][k];
            bH[e]  = BsH[br][k];      bA[e]  = BsA[br][k];
        }
        acc0 = __builtin_amdgcn_wmma_f32_16x16x32_bf16(false, aA0, false, bH, (short)0, acc0, false, false);
        acc0 = __builtin_amdgcn_wmma_f32_16x16x32_bf16(false, aH0, false, bA, (short)0, acc0, false, false);
        acc1 = __builtin_amdgcn_wmma_f32_16x16x32_bf16(false, aA1, false, bH, (short)0, acc1, false, false);
        acc1 = __builtin_amdgcn_wmma_f32_16x16x32_bf16(false, aH1, false, bA, (short)0, acc1, false, false);
        __syncthreads();
    }

    int n = n0 + wn * 16 + (lane & 15);
    int mb = m0 + wm * 32 + ((lane >> 4) ? 8 : 0);
#pragma unroll
    for (int r = 0; r < 8; ++r) {
        pE[(mb + r) * NN + n]      = acc0[r];
        pE[(mb + 16 + r) * NN + n] = acc1[r];
    }
}

// ---------------------------------------------------------------------------
// Fused mask + softmax (axis=1 == rows of symmetric E) + *adj -> bf16 S.
// ---------------------------------------------------------------------------
__global__ __launch_bounds__(256)
void softmax_mask_kernel(const float* __restrict__ E, const float* __restrict__ adj,
                         __bf16* __restrict__ S) {
    long long row = blockIdx.x;
    const float* er = E   + row * NN;
    const float* ar = adj + row * NN;
    __bf16* sr = S + row * NN;
    __shared__ float red[256];
    int tid = threadIdx.x;

    float ev[4], av[4];
    float mx = -3.0e38f;
#pragma unroll
    for (int it = 0; it < 4; ++it) {
        int i = tid + it * 256;
        av[it] = ar[i];
        ev[it] = er[i];
        mx = fmaxf(mx, (av[it] > 0.0f) ? ev[it] : -3.0e38f);
    }
    red[tid] = mx; __syncthreads();
    for (int s = 128; s; s >>= 1) { if (tid < s) red[tid] = fmaxf(red[tid], red[tid + s]); __syncthreads(); }
    mx = red[0]; __syncthreads();

    float p[4], sum = 0.0f;
#pragma unroll
    for (int it = 0; it < 4; ++it) {
        p[it] = (av[it] > 0.0f) ? __expf(ev[it] - mx) : 0.0f;
        sum += p[it];
    }
    red[tid] = sum; __syncthreads();
    for (int s = 128; s; s >>= 1) { if (tid < s) red[tid] += red[tid + s]; __syncthreads(); }
    float inv = 1.0f / red[0];

#pragma unroll
    for (int it = 0; it < 4; ++it)
        sr[tid + it * 256] = (__bf16)(p[it] * inv);
}

// ---------------------------------------------------------------------------
// Gated residual update: one wave per node (wave32).
// ---------------------------------------------------------------------------
__global__ __launch_bounds__(256)
void gate_update_kernel(float* __restrict__ h, const float* __restrict__ hp,
                        const float* __restrict__ gw, const float* __restrict__ gb,
                        __bf16* __restrict__ hb) {
    int wave = threadIdx.x >> 5, lane = threadIdx.x & 31;
    int node = blockIdx.x * 8 + wave;
    float* hr = h + node * DP;
    const float* hpr = hp + node * DP;
    float z = 0.0f;
    for (int k = lane; k < DD; k += 32)
        z += hr[k] * gw[k] + hpr[k] * gw[DD + k];
    for (int off = 16; off; off >>= 1) z += __shfl_down(z, off, 32);
    z = __shfl(z, 0, 32);
    float c = 1.0f / (1.0f + __expf(-(z + gb[0])));
    __bf16* hbr = hb + node * DP;
    for (int k = lane; k < DP; k += 32) {
        float nh = c * hr[k] + (1.0f - c) * hpr[k];
        hr[k] = nh;
        hbr[k] = (__bf16)nh;
    }
}

// ---------------------------------------------------------------------------
// Graph readout: hsum[b,k] = sum_n h[b,n,k]
// ---------------------------------------------------------------------------
__global__ void readout_kernel(const float* __restrict__ h, float* __restrict__ hsum) {
    int b = blockIdx.x, k = threadIdx.x;
    if (k >= DP) return;
    const float* p = h + (long long)b * NN * DP + k;
    float acc = 0.0f;
    for (int n = 0; n < NN; ++n) acc += p[n * DP];
    hsum[b * DP + k] = acc;
}

// ---------------------------------------------------------------------------
// Final MLP (tiny, M=16): one block per batch element.
// ---------------------------------------------------------------------------
__global__ __launch_bounds__(256)
void mlp_kernel(const float* __restrict__ hsum,
                const float* __restrict__ W0, const float* __restrict__ b0,
                const float* __restrict__ W1, const float* __restrict__ b1,
                const float* __restrict__ W2, const float* __restrict__ b2,
                const float* __restrict__ W3, const float* __restrict__ b3,
                float* __restrict__ out) {
    __shared__ float s0[256], s1[256];
    int b = blockIdx.x, t = threadIdx.x;
    if (t < DD) s0[t] = hsum[b * DP + t];
    __syncthreads();
    if (t < DFC_) {
        float v = b0[t];
        for (int k = 0; k < DD; ++k) v += s0[k] * W0[k * DFC_ + t];
        s1[t] = fmaxf(v, 0.0f);
    }
    __syncthreads();
    if (t < DFC_) {
        float v = b1[t];
        for (int k = 0; k < DFC_; ++k) v += s1[k] * W1[k * DFC_ + t];
        s0[t] = fmaxf(v, 0.0f);
    }
    __syncthreads();
    if (t < DFC_) {
        float v = b2[t];
        for (int k = 0; k < DFC_; ++k) v += s0[k] * W2[k * DFC_ + t];
        s1[t] = fmaxf(v, 0.0f);
    }
    __syncthreads();
    if (t < FINAL_) {
        float v = b3[t];
        for (int k = 0; k < DFC_; ++k) v += s1[k] * W3[k * FINAL_ + t];
        out[b * FINAL_ + t] = 1.0f / (1.0f + __expf(-v));
    }
}

// ---------------------------------------------------------------------------
extern "C" void kernel_launch(void* const* d_in, const int* in_sizes, int n_in,
                              void* d_out, int out_size, void* d_ws, size_t ws_size,
                              hipStream_t stream) {
    const float* x     = (const float*)d_in[0];
    const float* adj   = (const float*)d_in[1];
    const float* eW    = (const float*)d_in[2];
    const float* gW    = (const float*)d_in[3];
    const float* gb    = (const float*)d_in[4];
    const float* gA    = (const float*)d_in[5];
    const float* gateW = (const float*)d_in[6];
    const float* gateb = (const float*)d_in[7];
    const float* fc0W  = (const float*)d_in[8];
    const float* fc0b  = (const float*)d_in[9];
    const float* fc1W  = (const float*)d_in[10];
    const float* fc1b  = (const float*)d_in[11];
    const float* fc2W  = (const float*)d_in[12];
    const float* fc2b  = (const float*)d_in[13];
    const float* fc3W  = (const float*)d_in[14];
    const float* fc3b  = (const float*)d_in[15];
    float* out = (float*)d_out;

    char* w = (char*)d_ws;
    auto alloc = [&](size_t bytes) -> char* {
        char* p = w;
        w += (bytes + 255) & ~(size_t)255;
        return p;
    };
    float*  h     = (float*) alloc((size_t)BN * DP * 4);
    float*  hp    = (float*) alloc((size_t)BN * DP * 4);
    __bf16* hb    = (__bf16*)alloc((size_t)BN * DP * 2);
    __bf16* h2b   = (__bf16*)alloc((size_t)BN * DP * 2);
    __bf16* hAb   = (__bf16*)alloc((size_t)BN * DP * 2);
    __bf16* xb    = (__bf16*)alloc((size_t)BN * KE * 2);
    __bf16* EWt   = (__bf16*)alloc((size_t)DP * KE * 2);
    __bf16* Wt    = (__bf16*)alloc((size_t)DP * DP * 2);
    __bf16* At    = (__bf16*)alloc((size_t)DP * DP * 2);
    float*  biasP = (float*) alloc((size_t)DP * 4);
    float*  hsum  = (float*) alloc((size_t)BB * DP * 4);
    float*  esym  = (float*) alloc((size_t)BB * NN * NN * 4);
    __bf16* S     = (__bf16*)alloc((size_t)BB * NN * NN * 2);

    // --- embed: cast/pad x, transpose/pad embede_W, GEMM -> h (f32) + hb (bf16)
    {
        int ne = BN * KE;
        pad_cast_kernel<<<(ne + 255) / 256, 256, 0, stream>>>(x, xb, BN, FIN_, KE);
        int nw = DP * KE;
        pad_castT_kernel<<<(nw + 255) / 256, 256, 0, stream>>>(eW, EWt, FIN_, DD, DP, KE);
        dim3 g(DP / 32, BN / 128, 1);
        gemm_nt_kernel<false, false><<<g, 256, 0, stream>>>(xb, EWt, h, hb, nullptr,
                                                            BN, DP, KE, KE, KE, DP, 0,
                                                            0, 0, 0);
    }

    for (int l = 0; l < LL; ++l) {
        int nw = DP * DP;
        pad_castT_kernel<<<(nw + 255) / 256, 256, 0, stream>>>(gW + l * DD * DD, Wt, DD, DD, DP, DP);
        pad_castT_kernel<<<(nw + 255) / 256, 256, 0, stream>>>(gA + l * DD * DD, At, DD, DD, DP, DP);
        pad_bias_kernel<<<1, 256, 0, stream>>>(gb + l * DD, biasP, DD, DP);

        dim3 gS(DP / 32, BN / 128, 1);
        // h2 = h @ W + b   (bf16 out)
        gemm_nt_kernel<false, false><<<gS, 256, 0, stream>>>(hb, Wt, nullptr, h2b, biasP,
                                                             BN, DP, DP, DP, DP, DP, 0,
                                                             0, 0, 0);
        // hA = h2 @ A      (bf16 out)
        gemm_nt_kernel<false, false><<<gS, 256, 0, stream>>>(h2b, At, nullptr, hAb, nullptr,
                                                             BN, DP, DP, DP, DP, DP, 0,
                                                             0, 0, 0);
        // e_sym = hA.h2^T + h2.hA^T  (f32, per batch)
        {
            dim3 g(NN / 32, NN / 128, BB);
            gemm_sym_kernel<<<g, 256, 0, stream>>>(hAb, h2b, esym, DP);
        }
        // mask + softmax(axis=1 via symmetry) + *adj -> S (bf16)
        softmax_mask_kernel<<<BB * NN, 256, 0, stream>>>(esym, adj, S);
        // h' = relu(S^T @ h2)  : TN GEMM, batched over z
        {
            dim3 g(DP / 32, NN / 128, BB);
            gemm_nt_kernel<true, true><<<g, 256, 0, stream>>>(S, h2b, hp, nullptr, nullptr,
                                                              NN, DP, NN, NN, DP, DP, 1,
                                                              (long long)NN * NN,
                                                              (long long)NN * DP,
                                                              (long long)NN * DP);
        }
        // gated residual update -> h (f32) and hb (bf16)
        gate_update_kernel<<<BN / 8, 256, 0, stream>>>(h, hp,
                                                       gateW + l * 2 * DD,
                                                       gateb + l, hb);
    }

    readout_kernel<<<BB, 256, 0, stream>>>(h, hsum);
    mlp_kernel<<<BB, 256, 0, stream>>>(hsum, fc0W, fc0b, fc1W, fc1b,
                                       fc2W, fc2b, fc3W, fc3b, out);
}
